// DA_conv_21449066676687
// MI455X (gfx1250) — matrix operand order repaired and checked by reading, not compiled, not run on hardware
//
#include <hip/hip_runtime.h>
#include <hip/hip_bf16.h>

typedef __bf16 bf16_t;
typedef __attribute__((ext_vector_type(16))) __bf16 v16bf;
typedef __attribute__((ext_vector_type(8)))  __bf16 v8bf;
typedef __attribute__((ext_vector_type(8)))  float  v8f;

// ---------------------------------------------------------------------------
// Activation helper
// ---------------------------------------------------------------------------
template<int ACT>
__device__ __forceinline__ float actf(float x) {
    if constexpr (ACT == 1) {               // LeakyReLU(0.1)
        return x >= 0.f ? x : 0.1f * x;
    } else if constexpr (ACT == 2) {        // sigmoid
        return 1.f / (1.f + __expf(-x));
    } else {
        return x;
    }
}

// ---------------------------------------------------------------------------
// fp32 -> bf16 convert
// ---------------------------------------------------------------------------
__global__ void cvt_f32_bf16(const float* __restrict__ s, bf16_t* __restrict__ d, int n) {
    int i = blockIdx.x * blockDim.x + threadIdx.x;
    if (i < n) d[i] = (bf16_t)s[i];
}

// ---------------------------------------------------------------------------
// Tiled transpose (rows x cols, row-major) -> bf16 (cols x rows, row-major)
// blockDim = (32, 8); grid = (cols/32, rows/32)
// ---------------------------------------------------------------------------
template<typename T>
__global__ void transpose_to_bf16(const T* __restrict__ src, bf16_t* __restrict__ dst,
                                  int rows, int cols) {
    __shared__ float tile[32][33];
    int bx = blockIdx.x * 32, by = blockIdx.y * 32;
    #pragma unroll
    for (int j = 0; j < 32; j += 8)
        tile[threadIdx.y + j][threadIdx.x] =
            (float)src[(size_t)(by + threadIdx.y + j) * cols + bx + threadIdx.x];
    __syncthreads();
    #pragma unroll
    for (int j = 0; j < 32; j += 8)
        dst[(size_t)(bx + threadIdx.y + j) * rows + by + threadIdx.x] =
            (bf16_t)tile[threadIdx.x][threadIdx.y + j];
}

// ---------------------------------------------------------------------------
// Register-blocked bf16 WMMA GEMM:  C[m,n] = sum_k A[m,k] * B[n,k]   ("NT")
//   MODE 0: outB[m*ldo+n] = bf16(act(acc))            (row-major bf16)
//   MODE 1: outB[n*ldo+m] = bf16(act(acc))            (transposed bf16)
//   MODE 2: outF[m*ldo+n] = acc + bias[m] + degF[m*N+n]*attB[m*N+n]  (final)
// Wave tile 64x64 (4x4 fragments of 16x16); block = 8 waves = 128x256 tile.
// grid = (N/256, M/128)
// ---------------------------------------------------------------------------
template<int ACT, int MODE>
__global__ __launch_bounds__(256)
void gemm_nt_wmma(const bf16_t* __restrict__ A, int lda,
                  const bf16_t* __restrict__ B, int ldb,
                  int M, int N, int K,
                  float* __restrict__ outF, bf16_t* __restrict__ outB, int ldo,
                  const float* __restrict__ bias,
                  const float* __restrict__ degF,
                  const bf16_t* __restrict__ attB) {
    const int lane = threadIdx.x & 31;
    const int wave = threadIdx.x >> 5;
    const int lh   = lane >> 4;     // which 16-lane half
    const int lm   = lane & 15;
    const int wrow = wave >> 2;     // 2 wave-rows
    const int wcol = wave & 3;      // 4 wave-cols
    const int m0 = blockIdx.y * 128 + wrow * 64;
    const int n0 = blockIdx.x * 256 + wcol * 64;

    v8f acc[4][4] = {};

    // Per-lane row bases. A loaded here in contiguous k order: lane<16 holds K 0-15,
    // lane>=16 holds K 16-31 of row (m & 15).  B loaded in the same order: lane<16
    // holds K 0-15 of column lm, lane>=16 holds K 16-31 of column lm.
    const bf16_t* aRow[4];
    const bf16_t* bRow[4];
    #pragma unroll
    for (int fm = 0; fm < 4; ++fm) aRow[fm] = A + (size_t)(m0 + 16 * fm + lm) * lda;
    #pragma unroll
    for (int fn = 0; fn < 4; ++fn) bRow[fn] = B + (size_t)(n0 + 16 * fn + lm) * ldb;

    for (int kt = 0; kt < K; kt += 32) {
        if (kt + 256 < K) {   // stream next tiles toward L2/L0
            __builtin_prefetch(aRow[0] + kt + 256, 0, 1);
            __builtin_prefetch(bRow[0] + kt + 256, 0, 1);
        }
        v16bf a[4], b[4];
        #pragma unroll
        for (int fm = 0; fm < 4; ++fm) {
            a[fm] = *(const v16bf*)(aRow[fm] + kt + 16 * lh);        // K 0-15 / 16-31
        }
        #pragma unroll
        for (int fn = 0; fn < 4; ++fn)
            b[fn] = *(const v16bf*)(bRow[fn] + kt + 16 * lh);        // K 0-15 / 16-31
        #pragma unroll
        for (int fm = 0; fm < 4; ++fm)
            #pragma unroll
            for (int fn = 0; fn < 4; ++fn)
                acc[fm][fn] = __builtin_amdgcn_wmma_f32_16x16x32_bf16(
                    false, a[fm], false, b[fn], (short)0, acc[fm][fn], false, false);
    }

    // C/D layout: VGPR v -> (M = 8*lh + v, N = lm) within each 16x16 tile.
    #pragma unroll
    for (int fm = 0; fm < 4; ++fm) {
        #pragma unroll
        for (int fn = 0; fn < 4; ++fn) {
            if constexpr (MODE == 1) {
                const int n = n0 + 16 * fn + lm;
                const int mb = m0 + 16 * fm + 8 * lh;
                v8bf t;
                #pragma unroll
                for (int v = 0; v < 8; ++v) t[v] = (bf16_t)actf<ACT>(acc[fm][fn][v]);
                *(v8bf*)(outB + (size_t)n * ldo + mb) = t;   // 16B contiguous store
            } else {
                #pragma unroll
                for (int v = 0; v < 8; ++v) {
                    const int m = m0 + 16 * fm + 8 * lh + v;
                    const int n = n0 + 16 * fn + lm;
                    float x = actf<ACT>(acc[fm][fn][v]);
                    if constexpr (MODE == 2) {
                        const size_t idx = (size_t)m * N + n;
                        x += bias[m] + degF[idx] * (float)attB[idx];
                        outF[(size_t)m * ldo + n] = x;
                    } else {
                        outB[(size_t)m * ldo + n] = (bf16_t)x;
                    }
                }
            }
        }
    }
}

// ---------------------------------------------------------------------------
// kern[i][q] = sum_o T1[i][o] * W2[q][o]   (skinny GEMM, 9 outputs per row)
// ---------------------------------------------------------------------------
__global__ __launch_bounds__(256)
void kern_gen(const bf16_t* __restrict__ T1, const float* __restrict__ W2,
              float* __restrict__ kern) {
    __shared__ float red[256];
    const int i = blockIdx.x;
    const int t = threadIdx.x;
    float part[9];
    #pragma unroll
    for (int q = 0; q < 9; ++q) part[q] = 0.f;
    const bf16_t* row = T1 + (size_t)i * 4096;
    for (int o = t; o < 4096; o += 256) {
        const float v = (float)row[o];
        #pragma unroll
        for (int q = 0; q < 9; ++q) part[q] += v * W2[q * 4096 + o];
    }
    for (int q = 0; q < 9; ++q) {
        red[t] = part[q];
        __syncthreads();
        for (int s = 128; s > 0; s >>= 1) {
            if (t < s) red[t] += red[t + s];
            __syncthreads();
        }
        if (t == 0) kern[(size_t)i * 9 + q] = red[0];
        __syncthreads();
    }
}

// ---------------------------------------------------------------------------
// Depthwise 3x3 conv (pad 1, per-channel dynamic kernel) + leaky -> bf16
// ---------------------------------------------------------------------------
__global__ __launch_bounds__(256)
void dwconv3x3_leaky_bf16(const float* __restrict__ x, const float* __restrict__ kern,
                          bf16_t* __restrict__ out) {
    const int idx = blockIdx.x * 256 + threadIdx.x;     // 4096*4096 total
    const int ch = idx >> 12;
    const int p  = idx & 4095;
    const int y  = p >> 6, xx = p & 63;
    const float* kp   = kern + ch * 9;
    const float* base = x + (size_t)ch * 4096;
    float acc = 0.f;
    #pragma unroll
    for (int dy = 0; dy < 3; ++dy) {
        const int yy = y + dy - 1;
        if (yy < 0 || yy > 63) continue;
        #pragma unroll
        for (int dx = 0; dx < 3; ++dx) {
            const int xc = xx + dx - 1;
            if (xc < 0 || xc > 63) continue;
            acc += kp[dy * 3 + dx] * base[yy * 64 + xc];
        }
    }
    acc = acc >= 0.f ? acc : 0.1f * acc;
    out[idx] = (bf16_t)acc;
}

// ---------------------------------------------------------------------------
extern "C" void kernel_launch(void* const* d_in, const int* in_sizes, int n_in,
                              void* d_out, int out_size, void* d_ws, size_t ws_size,
                              hipStream_t stream) {
    const float* c   = (const float*)d_in[0];   // (1,4096,64,64)
    const float* deg = (const float*)d_in[1];   // (4096,4096)
    const float* W1  = (const float*)d_in[2];   // (4096,4096)
    const float* W2  = (const float*)d_in[3];   // (9,4096)
    const float* Wc  = (const float*)d_in[4];   // (4096,4096)
    const float* bc  = (const float*)d_in[5];   // (4096,)
    const float* Wd1 = (const float*)d_in[6];   // (512,4096)
    const float* Wd2 = (const float*)d_in[7];   // (4096,512)
    float* out = (float*)d_out;

    char* ws = (char*)d_ws;
    const size_t MB = 1ull << 20;
    bf16_t* degB = (bf16_t*)(ws + 0 * MB);      // 32MB; reused as convB after GEMM1
    bf16_t* degT = (bf16_t*)(ws + 32 * MB);     // 32MB
    bf16_t* W1B  = (bf16_t*)(ws + 64 * MB);     // 32MB; reused as ctB after GEMM1
    bf16_t* WcB  = (bf16_t*)(ws + 96 * MB);     // 32MB
    bf16_t* Wd1B = (bf16_t*)(ws + 128 * MB);    // 4MB
    bf16_t* Wd2B = (bf16_t*)(ws + 132 * MB);    // 4MB
    bf16_t* hidT = (bf16_t*)(ws + 136 * MB);    // 4MB
    bf16_t* T1B  = (bf16_t*)(ws + 140 * MB);    // 32MB; reused as attB after kern_gen
    float*  kern = (float*) (ws + 172 * MB);    // 144KB
    bf16_t* convB = degB;
    bf16_t* ctB   = W1B;
    bf16_t* attB  = T1B;

    const int n44 = 4096 * 4096;
    const int n45 = 512 * 4096;
    cvt_f32_bf16<<<n44 / 256, 256, 0, stream>>>(deg, degB, n44);
    cvt_f32_bf16<<<n44 / 256, 256, 0, stream>>>(W1,  W1B,  n44);
    cvt_f32_bf16<<<n44 / 256, 256, 0, stream>>>(Wc,  WcB,  n44);
    cvt_f32_bf16<<<n45 / 256, 256, 0, stream>>>(Wd1, Wd1B, n45);
    cvt_f32_bf16<<<n45 / 256, 256, 0, stream>>>(Wd2, Wd2B, n45);
    transpose_to_bf16<float><<<dim3(128, 128), dim3(32, 8), 0, stream>>>(deg, degT, 4096, 4096);

    // T1 = leaky(deg @ W1^T)   [4096x4096x4096 bf16 WMMA]
    gemm_nt_wmma<1, 0><<<dim3(16, 32), 256, 0, stream>>>(
        degB, 4096, W1B, 4096, 4096, 4096, 4096,
        nullptr, T1B, 4096, nullptr, nullptr, nullptr);

    // kern = T1 @ W2^T  -> (4096, 9) dynamic depthwise kernels
    kern_gen<<<4096, 256, 0, stream>>>(T1B, W2, kern);

    // depthwise conv + leaky  (writes bf16, channel-major)
    dwconv3x3_leaky_bf16<<<65536, 256, 0, stream>>>(c, kern, convB);

    // ct[p][c] = conv[c][p]  (so GEMM2 B-operand is K-contiguous)
    transpose_to_bf16<bf16_t><<<dim3(128, 128), dim3(32, 8), 0, stream>>>(convB, ctB, 4096, 4096);

    // hid^T[p][r] = leaky(sum_c Wd1[r][c] * deg[c][p])   [512x4096x4096]
    gemm_nt_wmma<1, 1><<<dim3(16, 4), 256, 0, stream>>>(
        Wd1B, 4096, degT, 4096, 512, 4096, 4096,
        nullptr, hidT, 512, nullptr, nullptr, nullptr);

    // att[o][p] = sigmoid(sum_r Wd2[o][r] * hid^T[p][r]) [4096x4096x512]
    gemm_nt_wmma<2, 0><<<dim3(16, 32), 256, 0, stream>>>(
        Wd2B, 512, hidT, 512, 4096, 4096, 512,
        nullptr, attB, 4096, nullptr, nullptr, nullptr);

    // out = Wc @ convL + bc + deg * att   [4096x4096x4096, fused epilogue]
    gemm_nt_wmma<0, 2><<<dim3(16, 32), 256, 0, stream>>>(
        WcB, 4096, ctB, 4096, 4096, 4096, 4096,
        out, nullptr, 4096, bc, deg, attB);
}
